// GTS_41188736368837
// MI455X (gfx1250) — matrix operand order repaired and checked by reading, not compiled, not run on hardware
//
#include <hip/hip_runtime.h>
#include <hip/hip_bf16.h>

// DCGRN cell on MI455X (gfx1250, wave32, WMMA bf16 16x16x32, fp32 accum).
// B=256, N=512, C=2, D=64, K=2 -> F=66, 3 diffusion taps.

#define Bn 256
#define Nn 512
#define Cc 2
#define Dd 64
#define Ff 66            // C + D
#define NC 16896         // Bn * Ff  (divisible by 16)
#define K3 198           // Ff * 3
#define KP 224           // K3 padded to multiple of 32
#define ND 32768         // Nn * Dd

typedef __attribute__((ext_vector_type(16))) __bf16 v16bf;
typedef __attribute__((ext_vector_type(8)))  __bf16 bf16x8;
typedef __attribute__((ext_vector_type(8)))  float   v8f;

union V16U { v16bf v; bf16x8 h[2]; };

// Build a 16-element bf16 WMMA operand from two contiguous 8-element runs.
static __device__ inline v16bf pack16(const __bf16* p0, const __bf16* p1) {
  V16U u;
  u.h[0] = *reinterpret_cast<const bf16x8*>(p0);
  u.h[1] = *reinterpret_cast<const bf16x8*>(p1);
  return u.v;
}

// CDNA5 async global->LDS copy (16B per lane), tracked by ASYNCcnt.
static __device__ inline void async_lds_b128(unsigned lds_byte_off, const __bf16* g) {
  asm volatile("global_load_async_to_lds_b128 %0, %1, off"
               :: "v"(lds_byte_off), "v"(g)
               : "memory");
}
static __device__ inline void wait_async0() {
  asm volatile("s_wait_asynccnt 0x0" ::: "memory");
}

// ---- 1. d_inv[j] = 1 / (1 + sum_k adj[j][k]) -------------------------------
__global__ void k_dinv(const float* __restrict__ adj, float* __restrict__ dinv) {
  int j = blockIdx.x * blockDim.x + threadIdx.x;
  if (j < Nn) {
    float s = 1.0f;  // +I contributes 1 on the diagonal of the row sum
    for (int k = 0; k < Nn; ++k) s += adj[j * Nn + k];
    dinv[j] = (s > 0.0f) ? (1.0f / s) : 0.0f;
  }
}

// ---- 2. adj_rw (transposed random walk) in bf16 ----------------------------
__global__ void k_adjbf(const float* __restrict__ adj, const float* __restrict__ dinv,
                        __bf16* __restrict__ Abf) {
  int idx = blockIdx.x * 256 + threadIdx.x;
  if (idx < Nn * Nn) {
    int i = idx / Nn, j = idx % Nn;
    float a = adj[j * Nn + i] + ((i == j) ? 1.0f : 0.0f);
    Abf[idx] = (__bf16)(a * dinv[j]);
  }
}

// ---- 3. pack X0 = [inputs|state] as (N, B*F) bf16 row-major ----------------
// One thread per packed u32 (two bf16). u32 index n*8448 + b*33 + f2 equals
// the flat thread index because NC/2 = Bn*33.
__global__ void k_pack0(const float* __restrict__ inputs, const float* __restrict__ state,
                        __bf16* __restrict__ X0) {
  int idx = blockIdx.x * 256 + threadIdx.x;   // over Nn*Bn*33 = 4.33M
  if (idx < Nn * Bn * 33) {
    int n  = idx / (Bn * 33);
    int r  = idx % (Bn * 33);
    int b  = r / 33;
    int f2 = r % 33;
    float v0, v1;
    if (f2 == 0) {
      v0 = inputs[b * (Nn * Cc) + n * Cc + 0];
      v1 = inputs[b * (Nn * Cc) + n * Cc + 1];
    } else {
      int f = 2 * f2 - 2;
      v0 = state[b * ND + n * Dd + f];
      v1 = state[b * ND + n * Dd + f + 1];
    }
    __bf16 p[2] = {(__bf16)v0, (__bf16)v1};
    reinterpret_cast<unsigned int*>(X0)[idx] = *reinterpret_cast<unsigned int*>(p);
  }
}

// ---- 4. repack weights: col-major bf16, K reordered k = tap*F + f, pad->KP -
__global__ void k_wpack(const float* __restrict__ Wg, const float* __restrict__ Wc,
                        __bf16* __restrict__ Wtg, __bf16* __restrict__ Wtc) {
  int idx = blockIdx.x * 256 + threadIdx.x;
  const int NG = 2 * Dd * KP;   // 128*224
  const int NCD = Dd * KP;      // 64*224
  if (idx < NG) {
    int o = idx / KP, k = idx % KP;
    float v = 0.0f;
    if (k < K3) { int kk = k / Ff, f = k % Ff; v = Wg[(f * 3 + kk) * (2 * Dd) + o]; }
    Wtg[idx] = (__bf16)v;
  } else if (idx < NG + NCD) {
    int id2 = idx - NG;
    int o = id2 / KP, k = id2 % KP;
    float v = 0.0f;
    if (k < K3) { int kk = k / Ff, f = k % Ff; v = Wc[(f * 3 + kk) * Dd + o]; }
    Wtc[id2] = (__bf16)v;
  }
}

// ---- 5/6/8/9. diffusion GEMM: Y = A(512x512) @ X(512xNC), bf16, fp32 accum.
// step2: Y = 2*(A@X) - X0prev (Chebyshev recurrence), fused epilogue.
// WG = 256 thr (8 waves) computes a 64x128 output block. Double-buffered LDS:
//  - A panel staged via global_load_async_to_lds_b128 (ASYNCcnt, no VGPR trip)
//  - X panel prefetched to registers, K-transposed into LDS after the WMMAs
//  - sched_barrier(0) pins the prefetch loads above the compute block
//  - one s_barrier per K-step (async wait precedes it for LDS visibility)
__global__ __launch_bounds__(256) void k_gemm_diff(
    const __bf16* __restrict__ A, const __bf16* __restrict__ X,
    const __bf16* __restrict__ X0prev, __bf16* __restrict__ Y, int step2) {
  __shared__ __bf16 lA[2][64 * 32];      // [m][k], pitch 32
  __shared__ __bf16 lX[2][128 * 32];     // [col][k] (K-transposed), pitch 32
  const int tid = threadIdx.x;
  const int wave = tid >> 5, lane = tid & 31;
  const int m_base = blockIdx.y * 64;
  const int c_base = blockIdx.x * 128;
  const int m_sub = wave >> 1;             // 0..3  (16-row subtile)
  const int c_sub = (wave & 1) * 64;       // 0/64  (wave owns 4 col subtiles)
  const int ar = tid >> 2, ac = (tid & 3) * 8;     // A staging role
  const int xr = tid >> 3, xc = (tid & 7) * 16;    // X staging role
  const int arow = m_sub * 16 + (lane & 15);
  const int koffA = (lane < 16) ? 0 : 8;           // ISA 16-bit A-operand layout
  const int bcol = lane & 15;
  const int koffB = (lane < 16) ? 0 : 8;           // ISA 16-bit B-operand layout
  const __bf16* gA = &A[(m_base + ar) * Nn + ac];
  const __bf16* gX = &X[(size_t)xr * NC + c_base + xc];
  const unsigned ldsA0 = (unsigned)(unsigned long long)&lA[0][ar * 32 + ac];
  const unsigned ldsA1 = (unsigned)(unsigned long long)&lA[1][ar * 32 + ac];
  v8f acc[4] = {};

  // Prologue: stage K-step 0 (A via async DMA, X via register transpose).
  async_lds_b128(ldsA0, gA);
  bf16x8 p0 = *reinterpret_cast<const bf16x8*>(gX);
  bf16x8 p1 = *reinterpret_cast<const bf16x8*>(gX + 8);
#pragma unroll
  for (int j = 0; j < 8; ++j) lX[0][(xc + j) * 32 + xr] = p0[j];
#pragma unroll
  for (int j = 0; j < 8; ++j) lX[0][(xc + 8 + j) * 32 + xr] = p1[j];
  wait_async0();
  __syncthreads();

  for (int ks = 0; ks < 16; ++ks) {
    const int cur = ks & 1, nxt = cur ^ 1;
    // Prefetch next panel (async A -> LDS, X -> registers), then fence the
    // scheduler so these issue before the compute block below.
    if (ks < 15) {
      const int k0 = (ks + 1) * 32;
      async_lds_b128(nxt ? ldsA1 : ldsA0, gA + k0);
      p0 = *reinterpret_cast<const bf16x8*>(gX + (size_t)k0 * NC);
      p1 = *reinterpret_cast<const bf16x8*>(gX + (size_t)k0 * NC + 8);
    }
    __builtin_amdgcn_sched_barrier(0);
    // Compute on the current buffers.
    v16bf av = pack16(&lA[cur][arow * 32 + koffA], &lA[cur][arow * 32 + 16 + koffA]);
#pragma unroll
    for (int j = 0; j < 4; ++j) {
      const __bf16* bp = &lX[cur][(c_sub + j * 16 + bcol) * 32 + koffB];
      v16bf bv = pack16(bp, bp + 16);
      acc[j] = __builtin_amdgcn_wmma_f32_16x16x32_bf16(
          false, av, false, bv, (short)0, acc[j], false, false);
    }
    // Store the prefetched X panel (transposed) and drain the async A copy.
    if (ks < 15) {
      *reinterpret_cast<bf16x8*>(&lX[nxt][0]);  // no-op alias anchor
#pragma unroll
      for (int j = 0; j < 8; ++j) lX[nxt][(xc + j) * 32 + xr] = p0[j];
#pragma unroll
      for (int j = 0; j < 8; ++j) lX[nxt][(xc + 8 + j) * 32 + xr] = p1[j];
      wait_async0();
    }
    __syncthreads();
  }

  // Epilogue: D-matrix layout -> lane holds one column, 8 rows per tile.
  const int rowoff = (lane >= 16) ? 8 : 0;
#pragma unroll
  for (int j = 0; j < 4; ++j) {
    int col = c_base + c_sub + j * 16 + bcol;
#pragma unroll
    for (int v = 0; v < 8; ++v) {
      int row = m_base + m_sub * 16 + rowoff + v;
      float val = acc[j][v];
      if (step2) val = 2.0f * val - (float)X0prev[(size_t)row * NC + col];
      Y[(size_t)row * NC + col] = (__bf16)val;
    }
  }
}

// Stage a wave's 16x224 feature rows [X0|X1|X2|0pad] into LDS.
// 48 jobs = 3 taps x 16 rows; each job copies a contiguous, 4B-aligned
// 132-byte run as 33 u32s (no per-element division).
static __device__ inline void stage_feat(__bf16* fw, const __bf16* const Xs[3],
                                         int b, int n0, int lane) {
  for (int j = lane; j < 48; j += 32) {
    const int kk = j >> 4, row = j & 15;
    const unsigned int* src = reinterpret_cast<const unsigned int*>(
        Xs[kk] + (size_t)(n0 + row) * NC + b * Ff);
    unsigned int* dst = reinterpret_cast<unsigned int*>(fw + row * KP + kk * Ff);
#pragma unroll
    for (int i = 0; i < 33; ++i) dst[i] = src[i];
  }
  if (lane < 16) {  // zero the 26-element K pad (13 u32s per row)
    unsigned int* dst = reinterpret_cast<unsigned int*>(fw + lane * KP + K3);
#pragma unroll
    for (int i = 0; i < 13; ++i) dst[i] = 0u;
  }
}

// ---- 7. gate: sigmoid([X0|X1|X2] @ Wg + bg); write u (fp32) and Xc = [in|r*state]
__global__ __launch_bounds__(256) void k_gate(
    const __bf16* __restrict__ X0, const __bf16* __restrict__ X1,
    const __bf16* __restrict__ X2, const __bf16* __restrict__ Wt,
    const float* __restrict__ bg, const float* __restrict__ inputs,
    const float* __restrict__ state, float* __restrict__ uws,
    __bf16* __restrict__ Xc) {
  __shared__ __bf16 feat[8][16 * KP];   // per-wave 16 rows x 224 features
  const int tid = threadIdx.x, wave = tid >> 5, lane = tid & 31;
  const int t = blockIdx.x * 8 + wave;  // 16-row tile id over (b, n)
  const int b = t >> 5;                 // 32 tiles per batch
  const int n0 = (t & 31) * 16;
  __bf16* fw = feat[wave];
  const __bf16* Xs[3] = {X0, X1, X2};
  stage_feat(fw, Xs, b, n0, lane);
  __syncthreads();

  v8f acc[8] = {};
  const int arow = lane & 15;
  const int koffA = (lane < 16) ? 0 : 8;
  const int bcol = lane & 15;
  const int koffB = (lane < 16) ? 0 : 8;
  for (int ks = 0; ks < 7; ++ks) {
    const __bf16* abase = &fw[arow * KP + ks * 32];
    v16bf av = pack16(abase + koffA, abase + 16 + koffA);
#pragma unroll
    for (int j = 0; j < 8; ++j) {
      const __bf16* bp = &Wt[(j * 16 + bcol) * KP + ks * 32 + koffB];
      v16bf bv = pack16(bp, bp + 16);
      acc[j] = __builtin_amdgcn_wmma_f32_16x16x32_bf16(
          false, av, false, bv, (short)0, acc[j], false, false);
    }
  }
  const int rowoff = (lane >= 16) ? 8 : 0;
#pragma unroll
  for (int j = 0; j < 8; ++j) {
    int o = j * 16 + bcol;
    float bias = bg[o];
#pragma unroll
    for (int v = 0; v < 8; ++v) {
      int node = n0 + rowoff + v;
      float g = 1.0f / (1.0f + __expf(-(acc[j][v] + bias)));
      if (o < Dd) {  // r-gate: fuse r*state into candidate input buffer
        float st = state[(size_t)b * ND + node * Dd + o];
        Xc[(size_t)node * NC + b * Ff + (Cc + o)] = (__bf16)(g * st);
      } else {       // u-gate
        uws[(size_t)b * ND + node * Dd + (o - Dd)] = g;
      }
    }
  }
  if (bcol < Cc) {  // copy the input features into Xc
#pragma unroll
    for (int v = 0; v < 8; ++v) {
      int node = n0 + rowoff + v;
      Xc[(size_t)node * NC + b * Ff + bcol] =
          (__bf16)inputs[(size_t)b * (Nn * Cc) + node * Cc + bcol];
    }
  }
}

// ---- 10. candidate + final blend: out = u*state + (1-u)*tanh(stk@Wc + bc) --
__global__ __launch_bounds__(256) void k_cand(
    const __bf16* __restrict__ X0, const __bf16* __restrict__ X1,
    const __bf16* __restrict__ X2, const __bf16* __restrict__ Wt,
    const float* __restrict__ bc, const float* __restrict__ state,
    const float* __restrict__ uws, float* __restrict__ out) {
  __shared__ __bf16 feat[8][16 * KP];
  const int tid = threadIdx.x, wave = tid >> 5, lane = tid & 31;
  const int t = blockIdx.x * 8 + wave;
  const int b = t >> 5;
  const int n0 = (t & 31) * 16;
  __bf16* fw = feat[wave];
  const __bf16* Xs[3] = {X0, X1, X2};
  stage_feat(fw, Xs, b, n0, lane);
  __syncthreads();

  v8f acc[4] = {};
  const int arow = lane & 15;
  const int koffA = (lane < 16) ? 0 : 8;
  const int bcol = lane & 15;
  const int koffB = (lane < 16) ? 0 : 8;
  for (int ks = 0; ks < 7; ++ks) {
    const __bf16* abase = &fw[arow * KP + ks * 32];
    v16bf av = pack16(abase + koffA, abase + 16 + koffA);
#pragma unroll
    for (int j = 0; j < 4; ++j) {
      const __bf16* bp = &Wt[(j * 16 + bcol) * KP + ks * 32 + koffB];
      v16bf bv = pack16(bp, bp + 16);
      acc[j] = __builtin_amdgcn_wmma_f32_16x16x32_bf16(
          false, av, false, bv, (short)0, acc[j], false, false);
    }
  }
  const int rowoff = (lane >= 16) ? 8 : 0;
#pragma unroll
  for (int j = 0; j < 4; ++j) {
    int o = j * 16 + bcol;
    float bias = bc[o];
#pragma unroll
    for (int v = 0; v < 8; ++v) {
      int node = n0 + rowoff + v;
      float c = tanhf(acc[j][v] + bias);
      size_t sidx = (size_t)b * ND + node * Dd + o;
      float u = uws[sidx];
      float st = state[sidx];
      out[sidx] = u * st + (1.0f - u) * c;
    }
  }
}

extern "C" void kernel_launch(void* const* d_in, const int* in_sizes, int n_in,
                              void* d_out, int out_size, void* d_ws, size_t ws_size,
                              hipStream_t stream) {
  const float* inputs = (const float*)d_in[0];
  const float* state  = (const float*)d_in[1];
  const float* adj    = (const float*)d_in[2];
  const float* Wg     = (const float*)d_in[3];
  const float* bg     = (const float*)d_in[4];
  const float* Wc     = (const float*)d_in[5];
  const float* bc     = (const float*)d_in[6];
  float* out = (float*)d_out;

  char* ws = (char*)d_ws;
  size_t off = 0;
  auto alloc = [&](size_t bytes) -> void* {
    void* p = ws + off;
    off += bytes;
    off = (off + 255) & ~(size_t)255;
    return p;
  };
  float*  dinv = (float*)alloc(Nn * sizeof(float));
  __bf16* Abf  = (__bf16*)alloc((size_t)Nn * Nn * 2);
  __bf16* Wtg  = (__bf16*)alloc((size_t)2 * Dd * KP * 2);
  __bf16* Wtc  = (__bf16*)alloc((size_t)Dd * KP * 2);
  const size_t SZX = (size_t)Nn * NC * 2;   // 16.5 MiB per X buffer
  __bf16* X0 = (__bf16*)alloc(SZX);
  __bf16* X1 = (__bf16*)alloc(SZX);
  __bf16* X2 = (__bf16*)alloc(SZX);
  __bf16* Xc = (__bf16*)alloc(SZX);
  float*  uws = (float*)alloc((size_t)Bn * ND * 4);

  k_dinv<<<2, 256, 0, stream>>>(adj, dinv);
  k_adjbf<<<(Nn * Nn + 255) / 256, 256, 0, stream>>>(adj, dinv, Abf);
  k_pack0<<<(Nn * Bn * 33 + 255) / 256, 256, 0, stream>>>(inputs, state, X0);
  k_wpack<<<((2 * Dd + Dd) * KP + 255) / 256, 256, 0, stream>>>(Wg, Wc, Wtg, Wtc);

  dim3 gg(NC / 128, Nn / 64);
  // Gate diffusion: X1 = A@X0 ; X2 = 2*A@X1 - X0
  k_gemm_diff<<<gg, 256, 0, stream>>>(Abf, X0, nullptr, X1, 0);
  k_gemm_diff<<<gg, 256, 0, stream>>>(Abf, X1, X0, X2, 1);
  k_gate<<<(Bn * Nn / 16) / 8, 256, 0, stream>>>(X0, X1, X2, Wtg, bg, inputs, state, uws, Xc);
  // Candidate diffusion (reuse X1/X2 buffers): X1 = A@Xc ; X2 = 2*A@X1 - Xc
  k_gemm_diff<<<gg, 256, 0, stream>>>(Abf, Xc, nullptr, X1, 0);
  k_gemm_diff<<<gg, 256, 0, stream>>>(Abf, X1, Xc, X2, 1);
  k_cand<<<(Bn * Nn / 16) / 8, 256, 0, stream>>>(Xc, X1, X2, Wtc, bc, state, uws, out);
}
